// HeteroGraphSAGE_52974126629631
// MI455X (gfx1250) — compile-verified, matrix operand order checked by reading, not run on hardware
//
#include <hip/hip_runtime.h>
#include <hip/hip_bf16.h>

typedef __attribute__((ext_vector_type(16))) _Float16 v16h;
typedef __attribute__((ext_vector_type(8)))  _Float16 v8h;
typedef __attribute__((ext_vector_type(2)))  _Float16 v2h;
typedef __attribute__((ext_vector_type(8)))  float    v8f;

#define N_NODES  100000
#define N_EDGES  1600000
#define N_FEAT   32
#define HIDDEN   128
#define N_GRAPHS 256
#define LN_EPS   1e-5f

// WMMA 16-bit A/B element -> K mapping (ISA 7.12.2):
// lanes 0-15 hold K-groups {0..7,16..23}; lanes 16-31 hold {8..15,24..31}
__device__ __forceinline__ int k_of(int lane, int i) {
    return (((lane >> 4) & 1) << 3) + (((i >> 3) & 1) << 4) + (i & 7);
}

// ---------------- utility kernels ----------------
__global__ void zero_f32(float* p, long long n) {
    long long i = (long long)blockIdx.x * blockDim.x + threadIdx.x;
    if (i < n) p[i] = 0.0f;
}

// Swizzle a row-major fp32 weight [Ktot x 128] into f16 WMMA B-tile layout.
// Tile (ktile, ntile) occupies 512 contiguous halves; slot = lane*16 + i.
__global__ void swizzle_w(const float* __restrict__ W, _Float16* __restrict__ Wsw, int Ktot) {
    int idx = blockIdx.x * blockDim.x + threadIdx.x;
    int total = (Ktot / 32) * 8 * 512;
    if (idx >= total) return;
    int b = idx >> 9;          // tile id = ktile*8 + ntile
    int s = idx & 511;
    int ktile = b >> 3;
    int ntile = b & 7;
    int lane = s >> 4;
    int i = s & 15;
    int k = ktile * 32 + k_of(lane, i);
    int n = ntile * 16 + (lane & 15);
    Wsw[idx] = (_Float16)W[(size_t)k * HIDDEN + n];
}

// ---------------- degree ----------------
__global__ void compute_deg(const int* __restrict__ dst, float* deg, int ne) {
    int e = blockIdx.x * blockDim.x + threadIdx.x;
    if (e < ne) atomicAdd(&deg[dst[e]], 1.0f);
}
__global__ void invert_deg(float* deg, int n) {
    int i = blockIdx.x * blockDim.x + threadIdx.x;
    if (i < n) deg[i] = 1.0f / fmaxf(deg[i], 1.0f);
}

// ---------------- input projection: h = x @ W_op + b_op ----------------
// One 16-row tile per block; 8 waves, each produces one 16x16 output tile (1 WMMA).
// A tile staged in LDS as f16 row-major (stride 40 halves = 80B, 16B-multiple);
// A operand = two ds_load_b128 per lane (row-major layout == WMMA A layout).
__global__ void input_proj(const float* __restrict__ x, const _Float16* __restrict__ Wsw,
                           const float* __restrict__ bop, float* __restrict__ h) {
    __shared__ __align__(16) _Float16 Ah[16 * 40];
    const int row0 = blockIdx.x * 16;
    const int tid = threadIdx.x;
    {   // stage 16x32 fp32 -> f16 pairs, coalesced (256 pairs, one per thread)
        int r = tid >> 4, cp = tid & 15;
        float2 v = *(const float2*)(x + (size_t)(row0 + r) * N_FEAT + 2 * cp);
        v2h p; p.x = (_Float16)v.x; p.y = (_Float16)v.y;
        *(v2h*)(Ah + r * 40 + 2 * cp) = p;
    }
    __syncthreads();
    const int w = tid >> 5, lane = tid & 31;
    const int hrow = lane & 15, mbase = (lane >> 4) * 8;
    const int koff = ((lane >> 4) & 1) * 8;
    union { v16h v; struct { v8h lo, hi; } p; } a;
    const _Float16* base = Ah + hrow * 40 + koff;
    a.p.lo = *(const v8h*)(base);        // K koff..koff+7
    a.p.hi = *(const v8h*)(base + 16);   // K koff+16..koff+23
    v16h b = *(const v16h*)(Wsw + (size_t)w * 512 + lane * 16);
    v8f acc = {};
    acc = __builtin_amdgcn_wmma_f32_16x16x32_f16(false, a.v, false, b, (short)0, acc, false, false);
    const int col = w * 16 + hrow;
    const float bias = bop[col];
    for (int r = 0; r < 8; ++r)
        h[(size_t)(row0 + mbase + r) * HIDDEN + col] = acc[r] + bias;
}

// ---------------- edge scatter: agg[dst] += h[src] (one wave per edge) ----------------
__global__ void scatter_edges(const float* __restrict__ h, const int* __restrict__ src,
                              const int* __restrict__ dst, float* __restrict__ agg, int ne) {
    long long gid = (long long)blockIdx.x * blockDim.x + threadIdx.x;
    int e = (int)(gid >> 5);
    int lane = (int)(gid & 31);
    if (e >= ne) return;
    int s = src[e], d = dst[e];
    const float4 v = *(const float4*)(h + (size_t)s * HIDDEN + lane * 4);
    float* o = agg + (size_t)d * HIDDEN + lane * 4;
    atomicAdd(o + 0, v.x); atomicAdd(o + 1, v.y);
    atomicAdd(o + 2, v.z); atomicAdd(o + 3, v.w);
}

// ---------------- fused SAGE layer: h = LN(ELU((agg*invdeg)@Wl + bl + h@Wr)) ----------------
// One 16-row tile per block, 8 waves = 8 N-tiles, 4 K-steps x 2 GEMMs = 8 WMMAs/wave.
// A tiles staged as f16 row-major (stride 136 halves = 272B); operand fetch is
// 2x ds_load_b128 per matrix per K-step (no scalar gathers/converts in hot loop).
__global__ void sage_layer(const float* __restrict__ agg, float* __restrict__ h,
                           const float* __restrict__ invdeg,
                           const _Float16* __restrict__ Wl, const _Float16* __restrict__ Wr,
                           const float* __restrict__ bl, const float* __restrict__ gamma,
                           const float* __restrict__ beta) {
    __shared__ __align__(16) _Float16 A1h[16 * 136];   // agg * invdeg (f16)
    __shared__ __align__(16) _Float16 A2h[16 * 136];   // h root (f16)
    __shared__ float Out[16 * 132];                    // ELU output for LayerNorm
    const int row0 = blockIdx.x * 16;
    const int tid = threadIdx.x;
    for (int it = 0; it < 4; ++it) {   // 1024 float2 pairs per matrix
        int p = it * 256 + tid;
        int r = p >> 6, cp = p & 63;
        size_t g = (size_t)(row0 + r) * HIDDEN + 2 * cp;
        float2 va = *(const float2*)(agg + g);
        float2 vh = *(const float2*)(h + g);
        float sc = invdeg[row0 + r];
        v2h pa, ph;
        pa.x = (_Float16)(va.x * sc); pa.y = (_Float16)(va.y * sc);
        ph.x = (_Float16)vh.x;        ph.y = (_Float16)vh.y;
        *(v2h*)(A1h + r * 136 + 2 * cp) = pa;
        *(v2h*)(A2h + r * 136 + 2 * cp) = ph;
    }
    __syncthreads();

    const int w = tid >> 5, lane = tid & 31;
    const int hrow = lane & 15, mbase = (lane >> 4) * 8;
    const int koff = ((lane >> 4) & 1) * 8;
    v8f acc = {};
    for (int kt = 0; kt < 4; ++kt) {
        union { v16h v; struct { v8h lo, hi; } p; } a1, a2;
        const _Float16* b1 = A1h + hrow * 136 + kt * 32 + koff;
        const _Float16* b2 = A2h + hrow * 136 + kt * 32 + koff;
        a1.p.lo = *(const v8h*)(b1);
        a1.p.hi = *(const v8h*)(b1 + 16);
        a2.p.lo = *(const v8h*)(b2);
        a2.p.hi = *(const v8h*)(b2 + 16);
        v16h w1 = *(const v16h*)(Wl + ((kt * 8 + w) * 512 + lane * 16));
        v16h w2 = *(const v16h*)(Wr + ((kt * 8 + w) * 512 + lane * 16));
        acc = __builtin_amdgcn_wmma_f32_16x16x32_f16(false, a1.v, false, w1, (short)0, acc, false, false);
        acc = __builtin_amdgcn_wmma_f32_16x16x32_f16(false, a2.v, false, w2, (short)0, acc, false, false);
    }

    const int col = w * 16 + hrow;
    const float bias = bl[col];
    for (int r = 0; r < 8; ++r) {
        float v = acc[r] + bias;
        v = (v > 0.0f) ? v : (expf(v) - 1.0f);    // ELU
        Out[(mbase + r) * 132 + col] = v;
    }
    __syncthreads();

    // LayerNorm: 16 rows x 16 threads/row (shuffle reduce within 16-lane groups)
    const int row = tid >> 4, t = tid & 15;
    float s = 0.0f;
    for (int j = 0; j < 8; ++j) s += Out[row * 132 + t + 16 * j];
    for (int o = 8; o >= 1; o >>= 1) s += __shfl_xor(s, o, 16);
    const float mu = s * (1.0f / HIDDEN);
    float vs = 0.0f;
    for (int j = 0; j < 8; ++j) {
        float d = Out[row * 132 + t + 16 * j] - mu;
        vs += d * d;
    }
    for (int o = 8; o >= 1; o >>= 1) vs += __shfl_xor(vs, o, 16);
    const float rstd = rsqrtf(vs * (1.0f / HIDDEN) + LN_EPS);
    for (int j = 0; j < 8; ++j) {
        int c = t + 16 * j;
        float d = (Out[row * 132 + c] - mu) * rstd * gamma[c] + beta[c];
        h[(size_t)(row0 + row) * HIDDEN + c] = d;  // in-place (own rows only)
    }
}

// ---------------- global mean pool + heads ----------------
__global__ void pool_scatter(const float* __restrict__ h, const int* __restrict__ batch,
                             float* gsum, float* gcnt, int n) {
    long long idx = (long long)blockIdx.x * blockDim.x + threadIdx.x;
    if (idx >= (long long)n * HIDDEN) return;
    int node = (int)(idx >> 7), f = (int)(idx & 127);
    int g = batch[node];
    atomicAdd(&gsum[g * HIDDEN + f], h[idx]);
    if (f == 0) atomicAdd(&gcnt[g], 1.0f);
}

__global__ void pool_finalize(const float* __restrict__ gsum, const float* __restrict__ gcnt,
                              const float* __restrict__ Wm, const float* __restrict__ bm,
                              const float* __restrict__ Wt, const float* __restrict__ bt,
                              float* __restrict__ out) {
    __shared__ float sm[HIDDEN];
    __shared__ float st[HIDDEN];
    const int g = blockIdx.x, tid = threadIdx.x;
    const float cnt = fmaxf(gcnt[g], 1.0f);
    const float p = gsum[g * HIDDEN + tid] / cnt;
    sm[tid] = p * Wm[tid];
    st[tid] = p * Wt[tid];
    __syncthreads();
    for (int s = HIDDEN / 2; s >= 1; s >>= 1) {
        if (tid < s) { sm[tid] += sm[tid + s]; st[tid] += st[tid + s]; }
        __syncthreads();
    }
    if (tid == 0) {
        out[g]            = sm[0] + bm[0];
        out[N_GRAPHS + g] = st[0] + bt[0];
    }
}

extern "C" void kernel_launch(void* const* d_in, const int* in_sizes, int n_in,
                              void* d_out, int out_size, void* d_ws, size_t ws_size,
                              hipStream_t stream) {
    const float* x      = (const float*)d_in[0];
    const int*   eidx   = (const int*)d_in[1];
    const int*   batch  = (const int*)d_in[2];
    const float* W_op   = (const float*)d_in[3];
    const float* b_op   = (const float*)d_in[4];
    const float* W_l    = (const float*)d_in[5];
    const float* b_l    = (const float*)d_in[6];
    const float* W_r    = (const float*)d_in[7];
    const float* gamma  = (const float*)d_in[8];
    const float* beta   = (const float*)d_in[9];
    const float* W_mem  = (const float*)d_in[10];
    const float* b_mem  = (const float*)d_in[11];
    const float* W_time = (const float*)d_in[12];
    const float* b_time = (const float*)d_in[13];
    const int* src = eidx;
    const int* dst = eidx + N_EDGES;
    float* out = (float*)d_out;

    // ---- workspace layout (256B aligned chunks) ----
    char* ws = (char*)d_ws;
    size_t off = 0;
    auto carve = [&](size_t bytes) {
        void* p = ws + off;
        off = (off + bytes + 255) & ~(size_t)255;
        return p;
    };
    float*    h     = (float*)carve((size_t)N_NODES * HIDDEN * sizeof(float));
    float*    agg   = (float*)carve((size_t)N_NODES * HIDDEN * sizeof(float));
    float*    ideg  = (float*)carve((size_t)N_NODES * sizeof(float));
    float*    gsum  = (float*)carve((size_t)N_GRAPHS * HIDDEN * sizeof(float));
    float*    gcnt  = (float*)carve((size_t)N_GRAPHS * sizeof(float));
    _Float16* Wop16 = (_Float16*)carve((size_t)N_FEAT * HIDDEN * sizeof(_Float16));
    _Float16* Wl16  = (_Float16*)carve((size_t)HIDDEN * HIDDEN * sizeof(_Float16));
    _Float16* Wr16  = (_Float16*)carve((size_t)HIDDEN * HIDDEN * sizeof(_Float16));
    (void)ws_size; (void)n_in; (void)in_sizes; (void)out_size;

    const int B = 256;
    const int nTiles = N_NODES / 16;  // 6250

    // weight swizzles (f16 WMMA B layout)
    swizzle_w<<<( (N_FEAT/32)*8*512 + B-1)/B, B, 0, stream>>>(W_op, Wop16, N_FEAT);
    swizzle_w<<<((HIDDEN/32)*8*512 + B-1)/B, B, 0, stream>>>(W_l, Wl16, HIDDEN);
    swizzle_w<<<((HIDDEN/32)*8*512 + B-1)/B, B, 0, stream>>>(W_r, Wr16, HIDDEN);

    // degrees
    zero_f32<<<(N_NODES + B-1)/B, B, 0, stream>>>(ideg, N_NODES);
    compute_deg<<<(N_EDGES + B-1)/B, B, 0, stream>>>(dst, ideg, N_EDGES);
    invert_deg<<<(N_NODES + B-1)/B, B, 0, stream>>>(ideg, N_NODES);

    // input projection
    input_proj<<<nTiles, B, 0, stream>>>(x, Wop16, b_op, h);

    // two SAGE layers
    const long long aggN = (long long)N_NODES * HIDDEN;
    for (int layer = 0; layer < 2; ++layer) {
        zero_f32<<<(int)((aggN + B-1)/B), B, 0, stream>>>(agg, aggN);
        scatter_edges<<<(int)(((long long)N_EDGES * 32 + B-1)/B), B, 0, stream>>>(h, src, dst, agg, N_EDGES);
        sage_layer<<<nTiles, B, 0, stream>>>(agg, h, ideg, Wl16, Wr16, b_l, gamma, beta);
    }

    // global mean pool + linear heads
    zero_f32<<<(N_GRAPHS * HIDDEN + N_GRAPHS + B-1)/B, B, 0, stream>>>(gsum, N_GRAPHS * HIDDEN + N_GRAPHS);
    pool_scatter<<<(int)((aggN + B-1)/B), B, 0, stream>>>(h, batch, gsum, gcnt, N_NODES);
    pool_finalize<<<N_GRAPHS, HIDDEN, 0, stream>>>(gsum, gcnt, W_mem, b_mem, W_time, b_time, out);
}
